// TransformerLayer_56719338111388
// MI455X (gfx1250) — matrix operand order repaired and checked by reading, not compiled, not run on hardware
//
#include <hip/hip_runtime.h>
#include <hip/hip_bf16.h>

// ---------------------------------------------------------------------------
// CDNA5 (gfx1250) transformer layer. wave32. All matrix math via
// V_WMMA_F32_16X16X32_F16. GEMMs use 128x128 block tiles: A and B staged in
// LDS with GLOBAL_LOAD_ASYNC_TO_LDS_B128 (ASYNCcnt), double-buffered,
// explicit 2x-unrolled pipeline (no buffer phi / fragment copies).
// ---------------------------------------------------------------------------

typedef _Float16 f16;
typedef __attribute__((ext_vector_type(16))) _Float16 v16h;
typedef __attribute__((ext_vector_type(8)))  _Float16 v8h;
typedef __attribute__((ext_vector_type(8)))  float    v8f;

#define D_MODEL 1024
#define D_HEAD  64
#define N_HEADS 16
#define D_HIDDEN 4096
#define SEQ 2048
#define BATCH 2
#define NTOK (BATCH*SEQ)   // 4096
#define BPITCH 40          // LDS row pitch (f16) for 32-wide K tiles

// ---- WMMA + fragment helpers (ISA 7.12.2 VGPR layouts) --------------------

__device__ __forceinline__ v8f wmma16(v16h a, v16h b, v8f c) {
  return __builtin_amdgcn_wmma_f32_16x16x32_f16(
      /*neg_a=*/false, a, /*neg_b=*/false, b,
      /*c_mod=*/(short)0, c, /*reuse_a=*/false, /*reuse_b=*/false);
}

// A fragment: A row-major [M,K]. Lane (r=lane&15, half=lane>>4) holds row r,
// K elems {half*8+0..7} then {16+half*8+0..7}.
__device__ __forceinline__ v16h load_a_frag(const f16* __restrict__ A, int ld, int lane) {
  const int half = lane >> 4, r = lane & 15;
  const f16* p = A + (size_t)r * ld + half * 8;
  v8h lo = *(const v8h*)(p);
  v8h hi = *(const v8h*)(p + 16);
  v16h out;
#pragma unroll
  for (int i = 0; i < 8; ++i) { out[i] = lo[i]; out[i + 8] = hi[i]; }
  return out;
}

// B fragment from BT row-major [N,K]: lane holds column n=r,
// K elems {half*8+0..7} then {16+half*8+0..7}.
__device__ __forceinline__ v16h load_b_frag(const f16* __restrict__ BT, int ld, int lane) {
  const int half = lane >> 4, r = lane & 15;
  const f16* p = BT + (size_t)r * ld + half * 8;
  v8h lo = *(const v8h*)(p);
  v8h hi = *(const v8h*)(p + 16);
  v16h out;
#pragma unroll
  for (int i = 0; i < 8; ++i) { out[i] = lo[i]; out[i + 8] = hi[i]; }
  return out;
}

// ---- CDNA5 async global->LDS copies (ASYNCcnt) ----------------------------
// Generic LDS pointer low 32 bits == LDS byte offset (ISA 10.2 aperture map).
__device__ __forceinline__ void async_copy16(f16* lds_dst, const f16* gsrc) {
  unsigned int l = (unsigned int)(size_t)(void*)lds_dst;
  asm volatile("global_load_async_to_lds_b128 %0, %1, off"
               :: "v"(l), "v"(gsrc) : "memory");
}
__device__ __forceinline__ void wait_async0() {
  asm volatile("s_wait_asynccnt 0x0" ::: "memory");
}
__device__ __forceinline__ void wait_async4() {
  asm volatile("s_wait_asynccnt 0x4" ::: "memory");
}

// Fill one 128(rows) x 32(K) tile into LDS (pitch BPITCH) from a row-major
// [rows, ld] source. 512 16B pieces / 256 threads = 2 async ops per thread.
__device__ __forceinline__ void fill128(f16* dst, const f16* __restrict__ src,
                                        int ld, int kk, int tid) {
#pragma unroll
  for (int p = 0; p < 2; ++p) {
    int piece = tid + p * 256;
    int row = piece >> 2, seg = piece & 3;
    async_copy16(dst + row * BPITCH + seg * 8,
                 src + (size_t)row * ld + kk + seg * 8);
  }
}

// One k-step for a wave's 64x32 sub-tile: 4 A frags, 2 B frags, 8 WMMAs
// issued back-to-back after a single dscnt drain.
__device__ __forceinline__ void compute_step(const f16* As, const f16* Bs,
                                             int wr, int wc, int lane, v8f acc[8]) {
  v16h a0 = load_a_frag(As + (wr * 64 +  0) * BPITCH, BPITCH, lane);
  v16h a1 = load_a_frag(As + (wr * 64 + 16) * BPITCH, BPITCH, lane);
  v16h a2 = load_a_frag(As + (wr * 64 + 32) * BPITCH, BPITCH, lane);
  v16h a3 = load_a_frag(As + (wr * 64 + 48) * BPITCH, BPITCH, lane);
  v16h b0 = load_b_frag(Bs + (wc * 32 +  0) * BPITCH, BPITCH, lane);
  v16h b1 = load_b_frag(Bs + (wc * 32 + 16) * BPITCH, BPITCH, lane);
  acc[0] = wmma16(a0, b0, acc[0]);
  acc[1] = wmma16(a0, b1, acc[1]);
  acc[2] = wmma16(a1, b0, acc[2]);
  acc[3] = wmma16(a1, b1, acc[3]);
  acc[4] = wmma16(a2, b0, acc[4]);
  acc[5] = wmma16(a2, b1, acc[5]);
  acc[6] = wmma16(a3, b0, acc[6]);
  acc[7] = wmma16(a3, b1, acc[7]);
}

// Shared 128x128 GEMM core. Kd must be a multiple of 64 (nk even).
__device__ __forceinline__ void gemm_core(const f16* __restrict__ Ag,
                                          const f16* __restrict__ Bg, int Kd,
                                          int wr, int wc, int lane, int tid,
                                          f16 (*As)[128 * BPITCH],
                                          f16 (*Bs)[128 * BPITCH], v8f acc[8]) {
  const int nk = Kd >> 5;
  fill128(As[0], Ag, Kd, 0, tid);
  fill128(Bs[0], Bg, Kd, 0, tid);
  for (int i = 0; i < nk; i += 2) {
    fill128(As[1], Ag, Kd, (i + 1) * 32, tid);
    fill128(Bs[1], Bg, Kd, (i + 1) * 32, tid);
    wait_async4();
    __syncthreads();
    compute_step(As[0], Bs[0], wr, wc, lane, acc);
    __syncthreads();
    if (i + 2 < nk) {
      fill128(As[0], Ag, Kd, (i + 2) * 32, tid);
      fill128(Bs[0], Bg, Kd, (i + 2) * 32, tid);
      wait_async4();
    } else {
      wait_async0();
    }
    __syncthreads();
    compute_step(As[1], Bs[1], wr, wc, lane, acc);
    __syncthreads();
  }
}

// ---- elementwise prep -----------------------------------------------------

__global__ void k_f32_to_f16(const float* __restrict__ src, f16* __restrict__ dst, int n) {
  int i = blockIdx.x * 256 + threadIdx.x;
  if (i < n) dst[i] = (f16)src[i];
}

// dst[n*K+k] = (f16)src[k*N+n]
__global__ void k_transpose_f16(const float* __restrict__ src, f16* __restrict__ dst,
                                int K, int N) {
  int i = blockIdx.x * 256 + threadIdx.x;
  if (i >= K * N) return;
  int n = i / K, k = i - n * K;
  dst[i] = (f16)src[(size_t)k * N + n];
}

// ---- fused QKV projection: block = 128 rows x 128-wide strip --------------
// Strips: 8 q, 8 k, 8 v. Q scaled 1/sqrt(64); V stored transposed [b,h,d,s].
__global__ void k_gemm_qkv(const f16* __restrict__ xh,
                           const f16* __restrict__ wqT, const f16* __restrict__ wkT,
                           const f16* __restrict__ wvT,
                           const float* __restrict__ bq, const float* __restrict__ bk,
                           const float* __restrict__ bv,
                           f16* __restrict__ Qh, f16* __restrict__ Kh, f16* __restrict__ Vt) {
  const int lane = threadIdx.x & 31, wave = threadIdx.x >> 5, tid = threadIdx.x;
  const int wr = wave >> 2, wc = wave & 3;
  const int bm = blockIdx.x / 24, ns = blockIdx.x % 24;
  const int mat = ns >> 3;                 // 0=q 1=k 2=v
  const int n0 = (ns & 7) * 128;
  const int m0 = bm * 128;
  const f16*  wT   = (mat == 0) ? wqT : (mat == 1) ? wkT : wvT;
  const float* bias = (mat == 0) ? bq  : (mat == 1) ? bk  : bv;

  __shared__ __attribute__((aligned(128))) f16 As[2][128 * BPITCH];
  __shared__ __attribute__((aligned(128))) f16 Bs[2][128 * BPITCH];
  v8f acc[8] = {};
  gemm_core(xh + (size_t)m0 * D_MODEL, wT + (size_t)n0 * D_MODEL, D_MODEL,
            wr, wc, lane, tid, As, Bs, acc);

  const int r = lane & 15, half = lane >> 4;
#pragma unroll
  for (int i = 0; i < 4; ++i)
#pragma unroll
    for (int jn = 0; jn < 2; ++jn)
#pragma unroll
      for (int j = 0; j < 8; ++j) {
        int tok = m0 + wr * 64 + i * 16 + j + 8 * half;
        int col = n0 + wc * 32 + jn * 16 + r;
        float v = acc[i * 2 + jn][j] + bias[col];
        int bi = tok >> 11, si = tok & (SEQ - 1);
        int h = col >> 6, d = col & 63;
        size_t bh = (size_t)bi * N_HEADS + h;
        if (mat == 0)       Qh[(bh * SEQ + si) * D_HEAD + d] = (f16)(v * 0.125f);
        else if (mat == 1)  Kh[(bh * SEQ + si) * D_HEAD + d] = (f16)v;
        else                Vt[(bh * D_HEAD + d) * SEQ + si] = (f16)v;
      }
}

// ---- flash-style causal attention ----------------------------------------
// One wave per (b,h, 16-query tile); consecutive waves share (b,h) for L2
// reuse. Q fragments hoisted; V loads issued before softmax VALU work.
__global__ void k_attention(const f16* __restrict__ Qh, const f16* __restrict__ Kh,
                            const f16* __restrict__ Vt, f16* __restrict__ ctx) {
  const int lane = threadIdx.x & 31, wave = threadIdx.x >> 5;
  const int gw = blockIdx.x * 8 + wave;
  const int qtile = gw & 127, bh = gw >> 7;
  const int q0 = qtile * 16;
  const f16* Qb = Qh + (size_t)bh * SEQ * D_HEAD;
  const f16* Kb = Kh + (size_t)bh * SEQ * D_HEAD;
  const f16* Vb = Vt + (size_t)bh * D_HEAD * SEQ;

  __shared__ __attribute__((aligned(64))) _Float16 Pl[8][16 * 32];
  _Float16* pbuf = &Pl[wave][0];

  const int r = lane & 15, half = lane >> 4;
  float st_m[8], st_l[8];
  v8f o[4] = {};
#pragma unroll
  for (int j = 0; j < 8; ++j) { st_m[j] = -1e30f; st_l[j] = 0.f; }

  // loop-invariant Q fragments
  v16h aq0 = load_a_frag(Qb + (size_t)q0 * D_HEAD +  0, D_HEAD, lane);
  v16h aq1 = load_a_frag(Qb + (size_t)q0 * D_HEAD + 32, D_HEAD, lane);

  const int nkb = qtile / 2 + 1;
  for (int kb = 0; kb < nkb; ++kb) {
    const int k0 = kb * 32;
    v8f s0 = {}, s1 = {};
    {
      v16h b00 = load_b_frag(Kb + (size_t)k0 * D_HEAD +  0, D_HEAD, lane);
      v16h b10 = load_b_frag(Kb + (size_t)(k0 + 16) * D_HEAD +  0, D_HEAD, lane);
      v16h b01 = load_b_frag(Kb + (size_t)k0 * D_HEAD + 32, D_HEAD, lane);
      v16h b11 = load_b_frag(Kb + (size_t)(k0 + 16) * D_HEAD + 32, D_HEAD, lane);
      s0 = wmma16(aq0, b00, s0); s0 = wmma16(aq1, b01, s0);
      s1 = wmma16(aq0, b10, s1); s1 = wmma16(aq1, b11, s1);
    }
    // start V loads early: latency hides under softmax VALU below
    v16h bv0 = load_b_frag(Vb + (size_t)( 0) * SEQ + k0, SEQ, lane);
    v16h bv1 = load_b_frag(Vb + (size_t)(16) * SEQ + k0, SEQ, lane);
    v16h bv2 = load_b_frag(Vb + (size_t)(32) * SEQ + k0, SEQ, lane);
    v16h bv3 = load_b_frag(Vb + (size_t)(48) * SEQ + k0, SEQ, lane);

    float p0[8], p1[8], alpha[8];
#pragma unroll
    for (int j = 0; j < 8; ++j) {
      int row = q0 + j + 8 * half;
      float v0 = (k0 + r      <= row) ? s0[j] : -1e30f;   // causal mask
      float v1 = (k0 + 16 + r <= row) ? s1[j] : -1e30f;
      float mx = fmaxf(v0, v1);
#pragma unroll
      for (int off = 8; off > 0; off >>= 1) mx = fmaxf(mx, __shfl_xor(mx, off, 16));
      float mnew = fmaxf(st_m[j], mx);
      float al = __expf(st_m[j] - mnew);
      float e0 = __expf(v0 - mnew), e1 = __expf(v1 - mnew);
      float rs = e0 + e1;
#pragma unroll
      for (int off = 8; off > 0; off >>= 1) rs += __shfl_xor(rs, off, 16);
      st_l[j] = st_l[j] * al + rs;
      st_m[j] = mnew;
      alpha[j] = al; p0[j] = e0; p1[j] = e1;
    }
#pragma unroll
    for (int j = 0; j < 8; ++j) {
      o[0][j] *= alpha[j]; o[1][j] *= alpha[j];
      o[2][j] *= alpha[j]; o[3][j] *= alpha[j];
    }
    // C-layout P -> LDS -> A-layout fragment (same-wave DS RAW)
#pragma unroll
    for (int j = 0; j < 8; ++j) {
      int lr = j + 8 * half;
      pbuf[lr * 32 + r]      = (_Float16)p0[j];
      pbuf[lr * 32 + 16 + r] = (_Float16)p1[j];
    }
    v16h ap = load_a_frag(pbuf, 32, lane);
    o[0] = wmma16(ap, bv0, o[0]);
    o[1] = wmma16(ap, bv1, o[1]);
    o[2] = wmma16(ap, bv2, o[2]);
    o[3] = wmma16(ap, bv3, o[3]);
  }
  const int bi = bh >> 4, h = bh & 15;
#pragma unroll
  for (int t = 0; t < 4; ++t) {
#pragma unroll
    for (int j = 0; j < 8; ++j) {
      int s = q0 + j + 8 * half;
      size_t tok = (size_t)bi * SEQ + s;
      int col = h * D_HEAD + t * 16 + r;
      ctx[tok * D_MODEL + col] = (f16)(o[t][j] / st_l[j]);
    }
  }
}

// ---- generic 128x128 GEMM, f32 out ----------------------------------------
__global__ void k_gemm_f32(const f16* __restrict__ A, const f16* __restrict__ BT,
                           float* __restrict__ C, int M, int N, int Kd) {
  const int lane = threadIdx.x & 31, wave = threadIdx.x >> 5, tid = threadIdx.x;
  const int wr = wave >> 2, wc = wave & 3;
  const int nblk = N >> 7;
  const int m0 = (blockIdx.x / nblk) * 128, n0 = (blockIdx.x % nblk) * 128;

  __shared__ __attribute__((aligned(128))) f16 As[2][128 * BPITCH];
  __shared__ __attribute__((aligned(128))) f16 Bs[2][128 * BPITCH];
  v8f acc[8] = {};
  gemm_core(A + (size_t)m0 * Kd, BT + (size_t)n0 * Kd, Kd,
            wr, wc, lane, tid, As, Bs, acc);

  const int r = lane & 15, half = lane >> 4;
#pragma unroll
  for (int i = 0; i < 4; ++i)
#pragma unroll
    for (int jn = 0; jn < 2; ++jn)
#pragma unroll
      for (int j = 0; j < 8; ++j) {
        int row = m0 + wr * 64 + i * 16 + j + 8 * half;
        int col = n0 + wc * 32 + jn * 16 + r;
        C[(size_t)row * N + col] = acc[i * 2 + jn][j];
      }
}

// ---- 128x128 GEMM + bias + exact GELU, f16 out ----------------------------
__global__ void k_gemm_gelu(const f16* __restrict__ A, const f16* __restrict__ BT,
                            const float* __restrict__ bias, f16* __restrict__ H,
                            int M, int N, int Kd) {
  const int lane = threadIdx.x & 31, wave = threadIdx.x >> 5, tid = threadIdx.x;
  const int wr = wave >> 2, wc = wave & 3;
  const int nblk = N >> 7;
  const int m0 = (blockIdx.x / nblk) * 128, n0 = (blockIdx.x % nblk) * 128;

  __shared__ __attribute__((aligned(128))) f16 As[2][128 * BPITCH];
  __shared__ __attribute__((aligned(128))) f16 Bs[2][128 * BPITCH];
  v8f acc[8] = {};
  gemm_core(A + (size_t)m0 * Kd, BT + (size_t)n0 * Kd, Kd,
            wr, wc, lane, tid, As, Bs, acc);

  const int r = lane & 15, half = lane >> 4;
#pragma unroll
  for (int i = 0; i < 4; ++i)
#pragma unroll
    for (int jn = 0; jn < 2; ++jn)
#pragma unroll
      for (int j = 0; j < 8; ++j) {
        int row = m0 + wr * 64 + i * 16 + j + 8 * half;
        int col = n0 + wc * 32 + jn * 16 + r;
        float v = acc[i * 2 + jn][j] + bias[col];
        float g = 0.5f * v * (1.f + erff(v * 0.70710678118f));   // exact GELU
        H[(size_t)row * N + col] = (f16)g;
      }
}

// ---- residual + LayerNorm (row per block) ---------------------------------
__global__ void k_add_ln(const float* __restrict__ resid, const float* __restrict__ y,
                         const float* __restrict__ g, const float* __restrict__ b,
                         float* __restrict__ out, f16* __restrict__ out_h) {
  const int row = blockIdx.x, tid = threadIdx.x;
  const float* yr = y + (size_t)row * D_MODEL;
  __shared__ float red[256];
  float s = 0.f;
  for (int i = tid; i < D_MODEL; i += 256) s += yr[i];
  red[tid] = s; __syncthreads();
  for (int st = 128; st > 0; st >>= 1) { if (tid < st) red[tid] += red[tid + st]; __syncthreads(); }
  float mu = red[0] * (1.f / D_MODEL);
  __syncthreads();
  float v = 0.f;
  for (int i = tid; i < D_MODEL; i += 256) { float d = yr[i] - mu; v += d * d; }
  red[tid] = v; __syncthreads();
  for (int st = 128; st > 0; st >>= 1) { if (tid < st) red[tid] += red[tid + st]; __syncthreads(); }
  float inv = rsqrtf(red[0] * (1.f / D_MODEL) + 1e-5f);
  for (int i = tid; i < D_MODEL; i += 256) {
    size_t idx = (size_t)row * D_MODEL + i;
    float o = resid[idx] + (yr[i] - mu) * inv * g[i] + b[i];
    out[idx] = o;
    if (out_h) out_h[idx] = (f16)o;
  }
}

// ---------------------------------------------------------------------------
extern "C" void kernel_launch(void* const* d_in, const int* in_sizes, int n_in,
                              void* d_out, int out_size, void* d_ws, size_t ws_size,
                              hipStream_t stream) {
  (void)in_sizes; (void)n_in; (void)out_size; (void)ws_size;
  const float* x     = (const float*)d_in[0];
  const float* wq    = (const float*)d_in[1];
  const float* bq    = (const float*)d_in[2];
  const float* wk    = (const float*)d_in[3];
  const float* bk    = (const float*)d_in[4];
  const float* wv    = (const float*)d_in[5];
  const float* bv    = (const float*)d_in[6];
  const float* wo    = (const float*)d_in[7];
  const float* ln1g  = (const float*)d_in[8];
  const float* ln1b  = (const float*)d_in[9];
  const float* w1    = (const float*)d_in[10];
  const float* b1    = (const float*)d_in[11];
  const float* w2    = (const float*)d_in[12];
  const float* ln2g  = (const float*)d_in[13];
  const float* ln2b  = (const float*)d_in[14];
  float* out = (float*)d_out;

  char* w = (char*)d_ws;
  auto alloc = [&](size_t bytes) -> char* {
    char* p = w; w += (bytes + 255) & ~(size_t)255; return p;
  };
  f16*  xh   = (f16*)alloc((size_t)NTOK * D_MODEL * 2);
  f16*  wqT  = (f16*)alloc((size_t)D_MODEL * D_MODEL * 2);
  f16*  wkT  = (f16*)alloc((size_t)D_MODEL * D_MODEL * 2);
  f16*  wvT  = (f16*)alloc((size_t)D_MODEL * D_MODEL * 2);
  f16*  woT  = (f16*)alloc((size_t)D_MODEL * D_MODEL * 2);
  f16*  w1T  = (f16*)alloc((size_t)D_MODEL * D_HIDDEN * 2);
  f16*  w2T  = (f16*)alloc((size_t)D_HIDDEN * D_MODEL * 2);
  f16*  Qh   = (f16*)alloc((size_t)NTOK * D_MODEL * 2);
  f16*  Kh   = (f16*)alloc((size_t)NTOK * D_MODEL * 2);
  f16*  Vt   = (f16*)alloc((size_t)NTOK * D_MODEL * 2);
  f16*  ctxh = (f16*)alloc((size_t)NTOK * D_MODEL * 2);
  float* y   = (float*)alloc((size_t)NTOK * D_MODEL * 4);
  float* x1  = (float*)alloc((size_t)NTOK * D_MODEL * 4);
  f16*  x1h  = (f16*)alloc((size_t)NTOK * D_MODEL * 2);
  f16*  hh   = (f16*)alloc((size_t)NTOK * D_HIDDEN * 2);
  float* z   = (float*)alloc((size_t)NTOK * D_MODEL * 4);

  const int EW = 256;
  k_f32_to_f16 <<<(NTOK * D_MODEL + EW - 1) / EW, EW, 0, stream>>>(x, xh, NTOK * D_MODEL);
  k_transpose_f16<<<(D_MODEL * D_MODEL + EW - 1) / EW, EW, 0, stream>>>(wq, wqT, D_MODEL, D_MODEL);
  k_transpose_f16<<<(D_MODEL * D_MODEL + EW - 1) / EW, EW, 0, stream>>>(wk, wkT, D_MODEL, D_MODEL);
  k_transpose_f16<<<(D_MODEL * D_MODEL + EW - 1) / EW, EW, 0, stream>>>(wv, wvT, D_MODEL, D_MODEL);
  k_transpose_f16<<<(D_MODEL * D_MODEL + EW - 1) / EW, EW, 0, stream>>>(wo, woT, D_MODEL, D_MODEL);
  k_transpose_f16<<<(D_MODEL * D_HIDDEN + EW - 1) / EW, EW, 0, stream>>>(w1, w1T, D_MODEL, D_HIDDEN);
  k_transpose_f16<<<(D_HIDDEN * D_MODEL + EW - 1) / EW, EW, 0, stream>>>(w2, w2T, D_HIDDEN, D_MODEL);

  // QKV: 32 m-blocks * 24 strips (8 each for q,k,v)
  k_gemm_qkv<<<32 * 24, 256, 0, stream>>>(xh, wqT, wkT, wvT, bq, bk, bv, Qh, Kh, Vt);
  // attention: 2*16*128 = 4096 waves
  k_attention<<<4096 / 8, 256, 0, stream>>>(Qh, Kh, Vt, ctxh);
  // out-proj
  k_gemm_f32<<<32 * 8, 256, 0, stream>>>(ctxh, woT, y, NTOK, D_MODEL, D_MODEL);
  k_add_ln<<<NTOK, 256, 0, stream>>>(x, y, ln1g, ln1b, x1, x1h);
  // FFN
  k_gemm_gelu<<<32 * 32, 256, 0, stream>>>(x1h, w1T, b1, hh, NTOK, D_HIDDEN, D_MODEL);
  k_gemm_f32<<<32 * 8, 256, 0, stream>>>(hh, w2T, z, NTOK, D_MODEL, D_HIDDEN);
  k_add_ln<<<NTOK, 256, 0, stream>>>(x1, z, ln2g, ln2b, out, nullptr);
}
